// CorrelationLayer_29351806501326
// MI455X (gfx1250) — matrix operand order repaired and checked by reading, not compiled, not run on hardware
//
#include <hip/hip_runtime.h>
#include <hip/hip_bf16.h>
#include <math.h>

typedef __attribute__((ext_vector_type(16))) _Float16 v16h;
typedef __attribute__((ext_vector_type(8)))  float    v8f;

#define BATCH 16
#define C_IN  256
#define FC    64
#define HGT   96
#define WID   96
#define HW    9216          // 96*96
#define K_CAT 320           // C_IN + FC
#define GP    40            // padded LDS pitch (halves): 80B rows -> conflict-free b128

// ---------------------------------------------------------------------------
// WMMA fragment helpers (CDNA5 wave32 layouts, CDNA5 ISA §7.12.2)
// ---------------------------------------------------------------------------

// A: 16x32 f16 from LDS row-major [row][pitch].
// lane<16: halves 0..7 = K0..7, 8..15 = K16..23 ; lane>=16: +8.
static __device__ __forceinline__ v16h load_a_frag(const _Float16* lds, int pitch,
                                                   int mtile, int lane) {
  int row   = mtile * 16 + (lane & 15);
  int kbase = (lane & 16) ? 8 : 0;
  const _Float16* p = lds + row * pitch + kbase;
  v16h f;
#pragma unroll
  for (int h = 0; h < 8; ++h) { f[h] = p[h]; f[h + 8] = p[h + 16]; }
  return f;
}

// B: 32x16 f16 from LDS stored TRANSPOSED [n][k] (pitch halves).
// lane<16: halves 0..7 = K0..7, 8..15 = K16..23, N=lane ; lane>=16: +8.
static __device__ __forceinline__ v16h load_bt_frag(const _Float16* lds, int pitch,
                                                    int ncol, int lane) {
  int n     = ncol + (lane & 15);
  int kbase = (lane & 16) ? 8 : 0;
  const _Float16* p = lds + n * pitch + kbase;
  v16h f;
#pragma unroll
  for (int h = 0; h < 8; ++h) { f[h] = p[h]; f[h + 8] = p[h + 16]; }
  return f;
}

static __device__ __forceinline__ v8f wmma_f16(v16h a, v16h b, v8f c) {
  return __builtin_amdgcn_wmma_f32_16x16x32_f16(false, a, false, b, (short)0, c,
                                                false, false);
}

// ---------------------------------------------------------------------------
// Prep kernels
// ---------------------------------------------------------------------------

__global__ __launch_bounds__(256) void corr_pool_kernel(const float* __restrict__ x,
                                                        float* __restrict__ pooled) {
  int bc = blockIdx.x;
  __shared__ float wsum[64];
  int t = threadIdx.x;
  if (t < 64) wsum[t] = 0.f;
  __syncthreads();
  const float* p = x + (size_t)bc * HW;
  for (int i = t; i < HW; i += 256) {
    int yy = i / WID, xx = i - yy * WID;
    atomicAdd(&wsum[(yy / 12) * 8 + (xx / 12)], p[i]);
  }
  __syncthreads();
  if (t < 64) pooled[(size_t)bc * 64 + t] = wsum[t] * (1.f / 144.f);
}

__global__ __launch_bounds__(256) void corr_kernorm_kernel(const float* __restrict__ pooled,
                                                           float* __restrict__ kerT) {
  int b = blockIdx.x >> 6, pidx = blockIdx.x & 63;
  int t = threadIdx.x;  // c
  float v = pooled[((size_t)(b * C_IN + t)) * 64 + pidx];
  __shared__ float red[256];
  __shared__ float scale;
  red[t] = v * v;
  __syncthreads();
  for (int off = 128; off > 0; off >>= 1) {
    if (t < off) red[t] += red[t + off];
    __syncthreads();
  }
  if (t == 0) scale = 1.0f / fmaxf(sqrtf(red[0]), 1e-12f);
  __syncthreads();
  kerT[((size_t)(b * FC + pidx)) * C_IN + t] = v * scale;
}

// M1[b][o][c] = sum_p W1[o][p] * kerT[b][p][c]  (fold 1x1 conv into corr kernel)
__global__ __launch_bounds__(256) void corr_m1_kernel(const float* __restrict__ W1,
                                                      const float* __restrict__ kerT,
                                                      _Float16* __restrict__ M1) {
  int b = blockIdx.x >> 6, o = blockIdx.x & 63;
  __shared__ float w[64];
  int t = threadIdx.x;  // c
  if (t < 64) w[t] = W1[o * 64 + t];
  __syncthreads();
  const float* kb = kerT + (size_t)b * FC * C_IN + t;
  float acc = 0.f;
#pragma unroll 8
  for (int p = 0; p < 64; ++p) acc += w[p] * kb[(size_t)p * C_IN];
  M1[((size_t)(b * FC + o)) * C_IN + t] = (_Float16)acc;
}

__global__ __launch_bounds__(256) void corr_invn_kernel(const float* __restrict__ x,
                                                        float* __restrict__ invn) {
  int i = blockIdx.x * 256 + threadIdx.x;  // B*HW = 147456 exactly
  int b = i / HW, hw = i - b * HW;
  const float* p = x + (size_t)b * C_IN * HW + hw;
  float s = 0.f;
  for (int c = 0; c < C_IN; ++c) {
    float v = p[(size_t)c * HW];
    s += v * v;
  }
  invn[i] = 1.0f / fmaxf(sqrtf(s), 1e-12f);
}

__global__ __launch_bounds__(256) void corr_wconv_kernel(const float* __restrict__ W2,
                                                         const float* __restrict__ W3,
                                                         _Float16* __restrict__ W2h,
                                                         _Float16* __restrict__ W3h) {
  int i = blockIdx.x * 256 + threadIdx.x;
  if (i < FC * FC * 9) {
    int tap = i % 9, oi = i / 9;
    int o = oi / FC, ii = oi - o * FC;
    W2h[(tap * FC + o) * FC + ii] = (_Float16)W2[i];
  }
  if (i < FC * K_CAT) W3h[i] = (_Float16)W3[i];
}

// ---------------------------------------------------------------------------
// GEMM1: hpre[b][o][hw] = sum_c M1[b][o][c] * (x[b][c][hw]*invn[b][hw]) + b1[o]
// M=64, K=256, N=9216.  256 threads (8 waves), tile M64 x N128, double-buffered.
// wave -> (mtile = wave&3, nhalf = wave>>2); 4 accumulators per wave.
// ---------------------------------------------------------------------------
__global__ __launch_bounds__(256) void corr_gemm1_kernel(const float* __restrict__ x,
                                                         const _Float16* __restrict__ M1,
                                                         const float* __restrict__ invn,
                                                         const float* __restrict__ b1,
                                                         float* __restrict__ hpre) {
  __shared__ __attribute__((aligned(16))) _Float16 Asl[2][64 * GP];
  __shared__ __attribute__((aligned(16))) _Float16 Bsl[2][128 * GP];
  __shared__ float invs[128];
  int b = blockIdx.y;
  int n0 = blockIdx.x * 128;
  int t = threadIdx.x, lane = t & 31, wave = t >> 5;
  int mtile = wave & 3, nhalf = wave >> 2;
  if (t < 128) invs[t] = invn[b * HW + n0 + t];
  const _Float16* Ab = M1 + (size_t)b * FC * C_IN;
  const float* xb = x + (size_t)b * C_IN * HW;
  v8f acc[4] = {};

  auto load_stage = [&](int k0, int s) {
    if (t < 128) {  // A: 64 rows x 32 halves
      int row = t >> 1, col = (t & 1) * 16;
      const _Float16* src = Ab + row * C_IN + k0 + col;
      _Float16* dst = Asl[s] + row * GP + col;
#pragma unroll
      for (int i = 0; i < 16; ++i) dst[i] = src[i];
    }
    {  // B: 32 k-rows x 128 cols, scaled x -> f16, stored transposed [n][k]
      int kr = t >> 3, seg = (t & 7) * 16;
      const float* src = xb + (size_t)(k0 + kr) * HW + n0 + seg;
      if (k0 + 32 < C_IN) __builtin_prefetch(src + (size_t)32 * HW, 0, 1);
      _Float16* dst = Bsl[s] + seg * GP + kr;
#pragma unroll
      for (int i = 0; i < 16; ++i) dst[i * GP] = (_Float16)(src[i] * invs[seg + i]);
    }
  };

  __syncthreads();  // invs visible
  load_stage(0, 0);
  const int KSTEPS = C_IN / 32;  // 8
  for (int kt = 0; kt < KSTEPS; ++kt) {
    int s = kt & 1;
    __syncthreads();  // stage s ready; stage s^1 consumed
    if (kt + 1 < KSTEPS) load_stage((kt + 1) * 32, s ^ 1);
    // batch all fragment loads, then back-to-back WMMAs
    v16h af = load_a_frag(Asl[s], GP, mtile, lane);
    v16h bf[4];
#pragma unroll
    for (int nsub = 0; nsub < 4; ++nsub)
      bf[nsub] = load_bt_frag(Bsl[s], GP, nhalf * 64 + nsub * 16, lane);
#pragma unroll
    for (int nsub = 0; nsub < 4; ++nsub)
      acc[nsub] = wmma_f16(af, bf[nsub], acc[nsub]);
  }
  int col = lane & 15, rbase = (lane & 16) ? 8 : 0;
  int hwbase = n0 + nhalf * 64;
#pragma unroll
  for (int nsub = 0; nsub < 4; ++nsub)
#pragma unroll
    for (int v = 0; v < 8; ++v) {
      int o = mtile * 16 + rbase + v;
      hpre[((size_t)(b * FC + o)) * HW + hwbase + nsub * 16 + col] = acc[nsub][v] + b1[o];
    }
}

// ---------------------------------------------------------------------------
// Instance-norm stats + apply
// ---------------------------------------------------------------------------
__global__ __launch_bounds__(256) void corr_stats_kernel(const float* __restrict__ src,
                                                         float* __restrict__ mv) {
  int bo = blockIdx.x, t = threadIdx.x;
  const float* p = src + (size_t)bo * HW;
  float s = 0.f, ss = 0.f;
  for (int i = t; i < HW; i += 256) {
    float v = p[i];
    s += v; ss += v * v;
  }
  __shared__ float rs[256], rss[256];
  rs[t] = s; rss[t] = ss;
  __syncthreads();
  for (int off = 128; off > 0; off >>= 1) {
    if (t < off) { rs[t] += rs[t + off]; rss[t] += rss[t + off]; }
    __syncthreads();
  }
  if (t == 0) {
    float mean = rs[0] * (1.f / HW);
    float var = rss[0] * (1.f / HW) - mean * mean;
    mv[bo * 2] = mean;
    mv[bo * 2 + 1] = rsqrtf(var + 1e-5f);
  }
}

__global__ __launch_bounds__(256) void corr_applyh_kernel(const float* __restrict__ hpre,
                                                          const float* __restrict__ mv,
                                                          _Float16* __restrict__ hN) {
  unsigned i = blockIdx.x * 256 + threadIdx.x;  // B*FC*HW
  int bo = i / HW;
  float v = (hpre[i] - mv[bo * 2]) * mv[bo * 2 + 1];
  hN[i] = (_Float16)(v > 0.f ? v : 0.f);
}

__global__ __launch_bounds__(256) void corr_applyout_kernel(const float* __restrict__ opre,
                                                            const float* __restrict__ mv,
                                                            float* __restrict__ out) {
  unsigned i = blockIdx.x * 256 + threadIdx.x;
  int bo = i / HW;
  float v = (opre[i] - mv[bo * 2]) * mv[bo * 2 + 1];
  out[i] = v > 0.f ? v : 0.f;
}

// ---------------------------------------------------------------------------
// 3x3 conv as 9 shifted WMMA GEMM accumulations.
// Block = (b, y), 192 threads (6 waves x 16-col subtiles => N=96), M=64, K=64.
// All 9 taps' A-slices preloaded per K-chunk -> 2 barriers per chunk.
// B tile stored transposed [ky][x][c] so every tap fragment is 2x ds_load_b128.
// ---------------------------------------------------------------------------
__global__ __launch_bounds__(192) void corr_conv3_kernel(const _Float16* __restrict__ hN,
                                                         const _Float16* __restrict__ W2h,
                                                         const float* __restrict__ b2,
                                                         _Float16* __restrict__ hc) {
  __shared__ __attribute__((aligned(16))) _Float16 Bs[3 * 98 * GP];  // [ky][x(-1..96)][ch]
  __shared__ __attribute__((aligned(16))) _Float16 As[9 * 64 * GP];  // [tap][o][i]
  int b = blockIdx.y, y = blockIdx.x;
  int t = threadIdx.x, lane = t & 31, wave = t >> 5;  // wave 0..5
  const _Float16* hb = hN + (size_t)b * FC * HW;
  v8f acc[4] = {};
  for (int chunk = 0; chunk < 2; ++chunk) {
    __syncthreads();  // previous chunk's compute done before overwrite
    if (t < 96) {     // B: (ky, c) segment across x, transposed scatter
      int r = t / 32, c = t - (t / 32) * 32;
      int yy = y - 1 + r;
      _Float16* dst = Bs + (r * 98) * GP + c;
      if (yy < 0 || yy >= HGT) {
        for (int i = 0; i < 98; ++i) dst[i * GP] = (_Float16)0.f;
      } else {
        const _Float16* src = hb + (size_t)(chunk * 32 + c) * HW + yy * WID;
        dst[0] = (_Float16)0.f;
        for (int i = 0; i < WID; ++i) dst[(1 + i) * GP] = src[i];
        dst[97 * GP] = (_Float16)0.f;
      }
    }
    if (t < 128) {    // A: all 9 taps, 64 rows x 32 halves each
      int row = t >> 1, col = (t & 1) * 16;
#pragma unroll
      for (int tap = 0; tap < 9; ++tap) {
        const _Float16* src = W2h + (size_t)(tap * FC + row) * FC + chunk * 32 + col;
        _Float16* dst = As + (tap * 64 + row) * GP + col;
#pragma unroll
        for (int i = 0; i < 16; ++i) dst[i] = src[i];
      }
    }
    __syncthreads();
#pragma unroll
    for (int tap = 0; tap < 9; ++tap) {
      int ky = tap / 3, kx = tap - ky * 3;
      // batch fragment loads, then back-to-back WMMAs
      v16h bf = load_bt_frag(Bs + (ky * 98) * GP, GP, wave * 16 + kx, lane);
      v16h af[4];
#pragma unroll
      for (int m = 0; m < 4; ++m)
        af[m] = load_a_frag(As + (tap * 64) * GP, GP, m, lane);
#pragma unroll
      for (int m = 0; m < 4; ++m)
        acc[m] = wmma_f16(af[m], bf, acc[m]);
    }
  }
  int col = lane & 15, rbase = (lane & 16) ? 8 : 0;
  int xc = wave * 16 + col;
#pragma unroll
  for (int m = 0; m < 4; ++m)
#pragma unroll
    for (int v = 0; v < 8; ++v) {
      int o = m * 16 + rbase + v;
      hc[((size_t)(b * FC + o)) * HW + y * WID + xc] = (_Float16)(acc[m][v] + b2[o]);
    }
}

// ---------------------------------------------------------------------------
// GEMM3: opre[b][o][hw] = sum_{c<256} W3[o][c]*x + sum_{j<64} W3[o][256+j]*hc + b3
// M=64, K=320, N=9216.  Same scheme as GEMM1 (double-buffered, N-tile 128).
// ---------------------------------------------------------------------------
__global__ __launch_bounds__(256) void corr_gemm3_kernel(const float* __restrict__ x,
                                                         const _Float16* __restrict__ W3h,
                                                         const _Float16* __restrict__ hc,
                                                         const float* __restrict__ b3,
                                                         float* __restrict__ opre) {
  __shared__ __attribute__((aligned(16))) _Float16 Asl[2][64 * GP];
  __shared__ __attribute__((aligned(16))) _Float16 Bsl[2][128 * GP];
  int b = blockIdx.y;
  int n0 = blockIdx.x * 128;
  int t = threadIdx.x, lane = t & 31, wave = t >> 5;
  int mtile = wave & 3, nhalf = wave >> 2;
  const float* xb = x + (size_t)b * C_IN * HW;
  const _Float16* hcb = hc + (size_t)b * FC * HW;
  v8f acc[4] = {};

  auto load_stage = [&](int k0, int s) {
    if (t < 128) {
      int row = t >> 1, col = (t & 1) * 16;
      const _Float16* src = W3h + row * K_CAT + k0 + col;
      _Float16* dst = Asl[s] + row * GP + col;
#pragma unroll
      for (int i = 0; i < 16; ++i) dst[i] = src[i];
    }
    {
      int kr = t >> 3, seg = (t & 7) * 16;
      _Float16* dst = Bsl[s] + seg * GP + kr;
      if (k0 < C_IN) {  // raw x channels -> f16
        const float* src = xb + (size_t)(k0 + kr) * HW + n0 + seg;
#pragma unroll
        for (int i = 0; i < 16; ++i) dst[i * GP] = (_Float16)src[i];
      } else {          // h channels already f16
        const _Float16* src = hcb + (size_t)(k0 - C_IN + kr) * HW + n0 + seg;
#pragma unroll
        for (int i = 0; i < 16; ++i) dst[i * GP] = src[i];
      }
    }
  };

  load_stage(0, 0);
  const int KSTEPS = K_CAT / 32;  // 10
  for (int kt = 0; kt < KSTEPS; ++kt) {
    int s = kt & 1;
    __syncthreads();
    if (kt + 1 < KSTEPS) load_stage((kt + 1) * 32, s ^ 1);
    v16h af = load_a_frag(Asl[s], GP, mtile, lane);
    v16h bf[4];
#pragma unroll
    for (int nsub = 0; nsub < 4; ++nsub)
      bf[nsub] = load_bt_frag(Bsl[s], GP, nhalf * 64 + nsub * 16, lane);
#pragma unroll
    for (int nsub = 0; nsub < 4; ++nsub)
      acc[nsub] = wmma_f16(af, bf[nsub], acc[nsub]);
  }
  int col = lane & 15, rbase = (lane & 16) ? 8 : 0;
  int hwbase = n0 + nhalf * 64;
#pragma unroll
  for (int nsub = 0; nsub < 4; ++nsub)
#pragma unroll
    for (int v = 0; v < 8; ++v) {
      int o = mtile * 16 + rbase + v;
      opre[((size_t)(b * FC + o)) * HW + hwbase + nsub * 16 + col] = acc[nsub][v] + b3[o];
    }
}

// ---------------------------------------------------------------------------
// Launch
// ---------------------------------------------------------------------------
extern "C" void kernel_launch(void* const* d_in, const int* in_sizes, int n_in,
                              void* d_out, int out_size, void* d_ws, size_t ws_size,
                              hipStream_t stream) {
  const float* x  = (const float*)d_in[0];
  const float* W1 = (const float*)d_in[1];
  const float* b1 = (const float*)d_in[2];
  const float* W2 = (const float*)d_in[3];
  const float* b2 = (const float*)d_in[4];
  const float* W3 = (const float*)d_in[5];
  const float* b3 = (const float*)d_in[6];
  float* out = (float*)d_out;

  char* w = (char*)d_ws;
  size_t off = 0;
  auto alloc = [&](size_t bytes) {
    void* p = w + off;
    off += (bytes + 255) & ~(size_t)255;
    return p;
  };
  float*    pooled = (float*)alloc((size_t)BATCH * C_IN * 64 * 4);        // 1 MB
  float*    kerT   = (float*)alloc((size_t)BATCH * FC * C_IN * 4);        // 1 MB
  _Float16* M1h    = (_Float16*)alloc((size_t)BATCH * FC * C_IN * 2);     // 0.5 MB
  float*    invn   = (float*)alloc((size_t)BATCH * HW * 4);               // 0.59 MB
  _Float16* W2h    = (_Float16*)alloc((size_t)9 * FC * FC * 2);           // 72 KB
  _Float16* W3h    = (_Float16*)alloc((size_t)FC * K_CAT * 2);            // 40 KB
  float*    hpre   = (float*)alloc((size_t)BATCH * FC * HW * 4);          // 37.7 MB
  float*    mv     = (float*)alloc((size_t)BATCH * FC * 2 * 4);           // 8 KB
  _Float16* hN     = (_Float16*)alloc((size_t)BATCH * FC * HW * 2);       // 18.9 MB
  _Float16* hc     = (_Float16*)alloc((size_t)BATCH * FC * HW * 2);       // 18.9 MB
  float*    opre   = hpre;  // hpre dead after applyh — reuse
  float*    mvo    = mv;    // mv1 dead after applyh — reuse

  const int nEl = BATCH * FC * HW;  // 9,437,184

  corr_wconv_kernel<<<dim3((FC * FC * 9 + 255) / 256), dim3(256), 0, stream>>>(W2, W3, W2h, W3h);
  corr_pool_kernel<<<dim3(BATCH * C_IN), dim3(256), 0, stream>>>(x, pooled);
  corr_kernorm_kernel<<<dim3(BATCH * 64), dim3(256), 0, stream>>>(pooled, kerT);
  corr_m1_kernel<<<dim3(BATCH * FC), dim3(256), 0, stream>>>(W1, kerT, M1h);
  corr_invn_kernel<<<dim3(BATCH * HW / 256), dim3(256), 0, stream>>>(x, invn);

  corr_gemm1_kernel<<<dim3(HW / 128, BATCH), dim3(256), 0, stream>>>(x, M1h, invn, b1, hpre);
  corr_stats_kernel<<<dim3(BATCH * FC), dim3(256), 0, stream>>>(hpre, mv);
  corr_applyh_kernel<<<dim3(nEl / 256), dim3(256), 0, stream>>>(hpre, mv, hN);

  corr_conv3_kernel<<<dim3(HGT, BATCH), dim3(192), 0, stream>>>(hN, W2h, b2, hc);

  corr_gemm3_kernel<<<dim3(HW / 128, BATCH), dim3(256), 0, stream>>>(x, W3h, hc, b3, opre);
  corr_stats_kernel<<<dim3(BATCH * FC), dim3(256), 0, stream>>>(opre, mvo);
  corr_applyout_kernel<<<dim3(nEl / 256), dim3(256), 0, stream>>>(opre, mvo, out);
  (void)in_sizes; (void)n_in; (void)out_size; (void)ws_size;
}
